// Update_31920196943969
// MI455X (gfx1250) — matrix operand order repaired and checked by reading, not compiled, not run on hardware
//
// MI455X (gfx1250, CDNA5) implementation of the GNN edge-update block.
//
// Strategy: ~24.5 GFLOP of GEMM over L2-resident data (192MB L2, 23.3TB/s
// HBM irrelevant after first touch) -> bf16 WMMA (v_wmma_f32_16x16x32_bf16)
// with fp32 accumulation is the throughput path on CDNA5. Weight tiles are
// staged to LDS with the Tensor Data Mover (tensor_load_to_lds +
// s_wait_tensorcnt, double buffered); activation fragments stream from L2
// with global b128 loads + global_prefetch. wave32 tiling: 1 wave = 16x64
// output strip (4 WMMA accumulators), 8 waves/block = 128x64 tile, K step 32.
// All 4 B fragments are loaded from LDS before the WMMA quartet so the four
// v_wmma issue back-to-back behind a single s_wait_dscnt.

#include <hip/hip_runtime.h>
#include <hip/hip_bf16.h>
#include <stdint.h>

#define DIMC      384
#define NEDGE     4096
#define KCORR     882
#define KCORR_PAD 896   // pad 882 -> 896 (multiple of 32) with zeros

typedef __attribute__((ext_vector_type(16))) __bf16 v16bf;
typedef __attribute__((ext_vector_type(8)))  __bf16 v8bf;
typedef __attribute__((ext_vector_type(8)))  float  v8f;
typedef __attribute__((ext_vector_type(4)))  unsigned int u32x4_t;
typedef __attribute__((ext_vector_type(8)))  int i32x8_t;
typedef __attribute__((ext_vector_type(4)))  int i32x4_t;

#if defined(__has_builtin)
#if __has_builtin(__builtin_amdgcn_tensor_load_to_lds)
#define HAVE_TDM 1
#endif
#endif
#ifndef HAVE_TDM
#define HAVE_TDM 0
#endif

__device__ __forceinline__ unsigned short f32_to_bf16_bits(float f) {
  unsigned u = __float_as_uint(f);
  unsigned rnd = 0x7fffu + ((u >> 16) & 1u);   // round-to-nearest-even
  return (unsigned short)((u + rnd) >> 16);
}

#if HAVE_TDM
// Build a 2D TDM descriptor per CDNA5 ISA and issue
// tensor_load_to_lds: tile = 64 rows x 32 bf16 (64B/row), row stride = ldw.
__device__ __forceinline__ void tdm_issue(const __bf16* gW, int ldw, int n0,
                                          int k, unsigned lds_off) {
  unsigned long long ga =
      (unsigned long long)(uintptr_t)(gW + (size_t)n0 * ldw + k);
  u32x4_t g0;
  g0[0] = 1u;                                   // count=1 (valid, user mode)
  g0[1] = lds_off;                              // lds_addr (bytes)
  g0[2] = (unsigned)ga;                         // global_addr[31:0]
  g0[3] = (unsigned)((ga >> 32) & 0x1FFFFFFull) // global_addr[56:32]
          | (2u << 30);                         // type=2 ("image")
  unsigned td0 = (unsigned)ldw;                 // tensor dim0 len (elements)
  unsigned td1 = 1u << 20;                      // generous dim1 bound
  i32x8_t g1;
  g1[0] = (int)(1u << 16);                      // data_size=1 (2B), mask=0
  g1[1] = (int)((td0 & 0xFFFFu) << 16);         // tensor_dim0[15:0]
  g1[2] = (int)((td0 >> 16) | ((td1 & 0xFFFFu) << 16));
  g1[3] = (int)((td1 >> 16) | (32u << 16));     // tile_dim0 = 32 elements
  g1[4] = 64;                                   // tile_dim1 = 64 rows
  g1[5] = (int)td0;                             // tensor_dim0_stride[31:0]
  g1[6] = 0;
  g1[7] = 0;
  i32x4_t gz = {0, 0, 0, 0};
#if __clang_major__ >= 23
  i32x8_t gz8 = {0, 0, 0, 0, 0, 0, 0, 0};
  __builtin_amdgcn_tensor_load_to_lds(g0, g1, gz, gz, gz8, 0);
#else
  __builtin_amdgcn_tensor_load_to_lds(g0, g1, gz, gz, 0);
#endif
}
#endif

// Y[M,384] = epi(X[M,K] @ W[384,K]^T + bias) (+ res). X,W bf16; Y fp32.
// epi: 0=none 1=relu 2=sigmoid 3=exp
__global__ __launch_bounds__(256) void k_gemm_wmma(
    const __bf16* __restrict__ X, int ldx, const __bf16* __restrict__ W,
    int ldw, const float* __restrict__ bias, const float* res, float* Y,
    int M, int K, int epi) {
  __shared__ __bf16 lw[2][64 * 32];            // double-buffered B tile, 8KB
  const int tid = threadIdx.x;
  const int wv = tid >> 5;                     // wave in block (0..7)
  const int lane = tid & 31;
  const int l15 = lane & 15;
  const int kh = lane >> 4;                    // K-half select (ISA layout)
  const int row0 = blockIdx.x * 128 + wv * 16; // this wave's 16 rows
  const int n0 = blockIdx.y * 64;              // 64 output cols

  v8f acc[4];
#pragma unroll
  for (int s = 0; s < 4; ++s) {
    float bv = bias[n0 + s * 16 + l15];        // C/D: N = lane%16
#pragma unroll
    for (int r = 0; r < 8; ++r) acc[s][r] = bv;
  }

  int ar = row0 + l15;                         // A: M = lane%16
  if (ar > M - 1) ar = M - 1;
  const __bf16* xrow = X + (size_t)ar * ldx;

  unsigned lds0 = (unsigned)(uintptr_t)(&lw[0][0]); // low 32b = LDS offset
  unsigned lds1 = (unsigned)(uintptr_t)(&lw[1][0]);
  (void)lds0; (void)lds1;

#if HAVE_TDM
  if (wv == 0) {                               // TDM ignores EXEC: uniform
    tdm_issue(W, ldw, n0, 0, lds0);            // per-wave branch, 1 issuer
    __builtin_amdgcn_s_wait_tensorcnt(0);
  }
#else
  for (int i = tid; i < 64 * 4; i += 256) {
    int rr = i >> 2, cg = i & 3;
    *(v8bf*)&lw[0][rr * 32 + cg * 8] =
        *(const v8bf*)(W + (size_t)(n0 + rr) * ldw + cg * 8);
  }
#endif
  __syncthreads();

  int buf = 0;
  for (int k = 0; k < K; k += 32) {
    int nk = k + 32;
    if (nk < K) {                              // prefetch next B tile
#if HAVE_TDM
      if (wv == 0) tdm_issue(W, ldw, n0, nk, buf ? lds0 : lds1);
#else
      for (int i = tid; i < 64 * 4; i += 256) {
        int rr = i >> 2, cg = i & 3;
        *(v8bf*)&lw[buf ^ 1][rr * 32 + cg * 8] =
            *(const v8bf*)(W + (size_t)(n0 + rr) * ldw + nk + cg * 8);
      }
#endif
      __builtin_prefetch((const void*)(xrow + nk), 0, 1); // global_prefetch
    }
    // A fragment: lane holds row l15; kh=0 -> K {0..7,16..23}, kh=1 -> +8
    v16bf a;
    {
      v8bf lo = *(const v8bf*)(xrow + k + kh * 8);
      v8bf hi = *(const v8bf*)(xrow + k + kh * 8 + 16);
#pragma unroll
      for (int i = 0; i < 8; ++i) { a[i] = lo[i]; a[8 + i] = hi[i]; }
    }
    // Load ALL four B fragments first (8 clustered ds_load_b128, one
    // s_wait_dscnt), then run the 4 WMMAs back-to-back: independent
    // accumulators -> no D->A/B RAW hazard between them (ISA 7.12.1).
    v16bf bm[4];
#pragma unroll
    for (int s = 0; s < 4; ++s) {
      // B fragment: lane (kh*16+n) holds W[n0+s*16+n][k + kh*8 .. +7, k + 16 + kh*8 .. +7]
      const __bf16* wp = &lw[buf][(s * 16 + l15) * 32 + kh * 8];
      v8bf b0 = *(const v8bf*)wp;
      v8bf b1 = *(const v8bf*)(wp + 16);
#pragma unroll
      for (int i = 0; i < 8; ++i) { bm[s][i] = b0[i]; bm[s][8 + i] = b1[i]; }
    }
#pragma unroll
    for (int s = 0; s < 4; ++s) {
      acc[s] = __builtin_amdgcn_wmma_f32_16x16x32_bf16(
          false, a, false, bm[s], (short)0, acc[s], false, false);
    }
#if HAVE_TDM
    if (wv == 0 && nk < K) __builtin_amdgcn_s_wait_tensorcnt(0);
#endif
    __syncthreads();
    buf ^= 1;
  }

#pragma unroll
  for (int s = 0; s < 4; ++s) {
    int col = n0 + s * 16 + l15;
#pragma unroll
    for (int r = 0; r < 8; ++r) {
      int row = row0 + r + 8 * kh;             // C/D: M = vgpr + 8*(lane/16)
      if (row < M) {
        float v = acc[s][r];
        if (epi == 1) v = fmaxf(v, 0.f);
        else if (epi == 2) v = 1.f / (1.f + __expf(-v));
        else if (epi == 3) v = __expf(v);
        if (res) v += res[(size_t)row * DIMC + col];
        Y[(size_t)row * DIMC + col] = v;
      }
    }
  }
}

// ---------- small support kernels ----------

__global__ void k_sum_ii(const int* ii, int n, float* out) {
  __shared__ float sh[256];
  float s = 0.f;
  for (int i = threadIdx.x; i < n; i += 256) s += (float)ii[i];
  sh[threadIdx.x] = s;
  __syncthreads();
  for (int st = 128; st > 0; st >>= 1) {
    if ((int)threadIdx.x < st) sh[threadIdx.x] += sh[threadIdx.x + st];
    __syncthreads();
  }
  if (threadIdx.x == 0) *out = sh[0] * 1e-10f;
}

__global__ void k_cvt_bf16(const float* src, unsigned short* dst, int rows,
                           int Ks, int Kd) {
  long i = (long)blockIdx.x * blockDim.x + threadIdx.x;
  long tot = (long)rows * Kd;
  if (i >= tot) return;
  int r = (int)(i / Kd), c = (int)(i % Kd);
  float v = (c < Ks) ? src[(size_t)r * Ks + c] : 0.f;   // zero-pad K
  dst[i] = f32_to_bf16_bits(v);
}

__global__ void k_add4(const float* a, const float* b, const float* c,
                       const float* bias, float* o, long n) {
  long i = (long)blockIdx.x * blockDim.x + threadIdx.x;
  if (i < n) o[i] = a[i] + b[i] + c[i] + *bias;
}

__global__ __launch_bounds__(128) void k_layernorm(const float* x,
                                                   const float* g,
                                                   const float* b, float* y,
                                                   int C, int relu) {
  int row = blockIdx.x;
  const float* xr = x + (size_t)row * C;
  float s = 0.f, s2 = 0.f;
  for (int c = threadIdx.x; c < C; c += 128) {
    float v = xr[c];
    s += v; s2 += v * v;
  }
  for (int o = 16; o > 0; o >>= 1) {
    s += __shfl_xor(s, o, 32);
    s2 += __shfl_xor(s2, o, 32);
  }
  __shared__ float ls[4], ls2[4];
  int w = threadIdx.x >> 5, ln = threadIdx.x & 31;
  if (ln == 0) { ls[w] = s; ls2[w] = s2; }
  __syncthreads();
  if (threadIdx.x == 0) {
    float a = 0.f, a2 = 0.f;
    for (int i = 0; i < 4; ++i) { a += ls[i]; a2 += ls2[i]; }
    ls[0] = a; ls2[0] = a2;
  }
  __syncthreads();
  float m = ls[0] / C;
  float var = ls2[0] / C - m * m;               // biased var (matches jnp.var)
  float inv = rsqrtf(var + 1e-3f);
  float* yr = y + (size_t)row * C;
  for (int c = threadIdx.x; c < C; c += 128) {
    float v = (xr[c] - m) * inv * g[c] + b[c];
    if (relu) v = fmaxf(v, 0.f);
    yr[c] = v;
  }
}

// argmax/argmin neighbor search, first-index tie-break (matches jnp.arg*).
__global__ void k_neighbors(const int* kk, const int* jj, int* ix, int* jx,
                            int n) {
  int t = blockIdx.x * blockDim.x + threadIdx.x;
  if (t >= n) return;
  int kn = kk[t], jn = jj[t];
  int bi = 0, bp = 0, bj = 0, bn = n;
  for (int m = 0; m < n; ++m) {
    if (kk[m] == kn) {
      int jm = jj[m];
      if (jm < jn && jm > bp) { bp = jm; bi = m; }
      if (jm > jn && jm < bn) { bn = jm; bj = m; }
    }
  }
  ix[t] = bi;
  jx[t] = bj;
}

__global__ void k_gather_rows(const float* src, const int* idx, float* dst,
                              long n, int C) {
  long i = (long)blockIdx.x * blockDim.x + threadIdx.x;
  if (i >= n) return;
  int r = (int)(i / C), c = (int)(i % C);
  dst[i] = src[(size_t)idx[r] * C + c];
}

__global__ void k_make_key(const int* ii, const int* jj, int* key, int n) {
  int t = blockIdx.x * blockDim.x + threadIdx.x;
  if (t >= n) return;
  int v = ii[t];
  v = v < 0 ? 0 : (v > 4095 ? 4095 : v);        // clip(ii,0,H-1)
  key[t] = v * 16 + jj[t];                      // same partition as ii*12345+jj
}

// Deterministic scatter-softmax aggregation: gout[g] = seg(fx*eg)/seg(eg).
__global__ __launch_bounds__(128) void k_segagg(const float* fx,
                                                const float* eg,
                                                const int* key, float* gout,
                                                int N, int C) {
  int g = blockIdx.x;
  int c = blockIdx.y * blockDim.x + threadIdx.x;
  if (c >= C) return;
  float den = 0.f, num = 0.f;
  for (int n = 0; n < N; ++n) {
    if (key[n] == g) {
      float e = eg[(size_t)n * C + c];
      den += e;
      num += fx[(size_t)n * C + c] * e;
    }
  }
  gout[(size_t)g * C + c] = (den > 0.f) ? num / den : 0.f;
}

__global__ void k_gather_add(float* h, const float* gsrc, const int* key,
                             long n, int C) {
  long i = (long)blockIdx.x * blockDim.x + threadIdx.x;
  if (i >= n) return;
  int r = (int)(i / C), c = (int)(i % C);
  h[i] += gsrc[(size_t)key[r] * C + c];
}

__global__ void k_grcomb(const float* x, const float* gate, const float* r,
                         float* o, long n) {
  long i = (long)blockIdx.x * blockDim.x + threadIdx.x;
  if (i < n) o[i] = x[i] + gate[i] * r[i];      // gate already sigmoided
}

__global__ __launch_bounds__(128) void k_heads(const float* h, const float* Wd,
                                               const float* bd,
                                               const float* Ww,
                                               const float* bw, const int* ii,
                                               float* outd, float* outw) {
  int row = blockIdx.x;
  const float* hr = h + (size_t)row * DIMC;
  float d0 = 0.f, d1 = 0.f, w0 = 0.f, w1 = 0.f;
  for (int c = threadIdx.x; c < DIMC; c += 128) {
    float v = fmaxf(hr[c], 0.f);
    d0 += v * Wd[c]; d1 += v * Wd[DIMC + c];
    w0 += v * Ww[c]; w1 += v * Ww[DIMC + c];
  }
  for (int o = 16; o > 0; o >>= 1) {
    d0 += __shfl_xor(d0, o, 32); d1 += __shfl_xor(d1, o, 32);
    w0 += __shfl_xor(w0, o, 32); w1 += __shfl_xor(w1, o, 32);
  }
  __shared__ float sd[4][4];
  int w = threadIdx.x >> 5, ln = threadIdx.x & 31;
  if (ln == 0) { sd[w][0] = d0; sd[w][1] = d1; sd[w][2] = w0; sd[w][3] = w1; }
  __syncthreads();
  if (threadIdx.x == 0) {
    float a0 = 0.f, a1 = 0.f, a2 = 0.f, a3 = 0.f;
    for (int i = 0; i < 4; ++i) {
      a0 += sd[i][0]; a1 += sd[i][1]; a2 += sd[i][2]; a3 += sd[i][3];
    }
    float b2 = (float)ii[row] * 1e-10f;
    outd[row * 2 + 0] = a0 + bd[0] + b2;
    outd[row * 2 + 1] = a1 + bd[1] + b2;
    outw[row * 2 + 0] = 1.f / (1.f + __expf(-(a2 + bw[0]))) + b2;
    outw[row * 2 + 1] = 1.f / (1.f + __expf(-(a3 + bw[1]))) + b2;
  }
}

// ---------- host orchestration ----------
// Input order: setup_inputs() insertion order; params flattened as a jax
// pytree (dict keys sorted alphabetically, recursively).
enum {
  IN_NET = 0, IN_INP, IN_CORR, IN_FLOW, IN_II, IN_JJ, IN_KK,
  P_Wd = 7, P_Ww,
  P_aij_Wf, P_aij_Wg, P_aij_Wh, P_aij_bf, P_aij_bg, P_aij_bh,
  P_akk_Wf, P_akk_Wg, P_akk_Wh, P_akk_bf, P_akk_bg, P_akk_bh,
  P_bd, P_bw,
  P_c1_W1, P_c1_W2, P_c1_b1, P_c1_b2,
  P_c2_W1, P_c2_W2, P_c2_b1, P_c2_b2,
  P_co_W1, P_co_W2, P_co_W3, P_co_b1, P_co_b2, P_co_b3, P_co_lnb, P_co_lng,
  P_g1_Wg, P_g1_bg, P_g1_W1, P_g1_W2, P_g1_b1, P_g1_b2,
  P_g2_Wg, P_g2_bg, P_g2_W1, P_g2_W2, P_g2_b1, P_g2_b2,
  P_ln1b, P_ln1g, P_ln2b, P_ln2g, P_nb, P_ng
};

extern "C" void kernel_launch(void* const* d_in, const int* in_sizes, int n_in,
                              void* d_out, int out_size, void* d_ws,
                              size_t ws_size, hipStream_t stream) {
  (void)in_sizes; (void)n_in; (void)out_size; (void)ws_size;
  const float* net = (const float*)d_in[IN_NET];
  const float* inp = (const float*)d_in[IN_INP];
  const float* corr = (const float*)d_in[IN_CORR];
  const int* iip = (const int*)d_in[IN_II];
  const int* jjp = (const int*)d_in[IN_JJ];
  const int* kkp = (const int*)d_in[IN_KK];
  auto F = [&](int i) { return (const float*)d_in[i]; };

  // ws carve (~34 MB)
  char* base = (char*)d_ws;
  size_t off = 0;
  auto carve = [&](size_t bytes) -> void* {
    void* p = base + off;
    off += (bytes + 255) & ~(size_t)255;
    return p;
  };
  float* s_bias = (float*)carve(sizeof(float));
  int* ixb = (int*)carve(NEDGE * 4);
  int* jxb = (int*)carve(NEDGE * 4);
  int* keyb = (int*)carve(NEDGE * 4);
  unsigned short* Xb = (unsigned short*)carve((size_t)NEDGE * KCORR_PAD * 2);
  unsigned short* Wb = (unsigned short*)carve((size_t)DIMC * KCORR_PAD * 2);
  unsigned short* Gb = (unsigned short*)carve((size_t)768 * DIMC * 2);
  float* t0 = (float*)carve((size_t)NEDGE * DIMC * 4);
  float* t1 = (float*)carve((size_t)NEDGE * DIMC * 4);
  float* t2 = (float*)carve((size_t)NEDGE * DIMC * 4);
  float* h = (float*)carve((size_t)NEDGE * DIMC * 4);
  float* gagg = (float*)carve((size_t)768 * DIMC * 4);

  const long NEL = (long)NEDGE * DIMC;
  dim3 EL((unsigned)((NEL + 255) / 256));

  auto cvt = [&](const float* src, unsigned short* dst, int rows, int Ks,
                 int Kd) {
    long tot = (long)rows * Kd;
    k_cvt_bf16<<<dim3((unsigned)((tot + 255) / 256)), 256, 0, stream>>>(
        src, dst, rows, Ks, Kd);
  };
  auto gemm = [&](const unsigned short* Xq, int ldx, const unsigned short* Wq,
                  int ldw, const float* bias, const float* res, float* Y,
                  int M, int K, int epi) {
    dim3 g((unsigned)(M / 128), DIMC / 64);
    k_gemm_wmma<<<g, 256, 0, stream>>>((const __bf16*)Xq, ldx,
                                       (const __bf16*)Wq, ldw, bias, res, Y, M,
                                       K, epi);
  };
  auto lnorm = [&](const float* x, const float* g, const float* b, float* y,
                   int relu) {
    k_layernorm<<<NEDGE, 128, 0, stream>>>(x, g, b, y, DIMC, relu);
  };
  auto soft_agg = [&](const int* key, int G, int iWf, int iWg, int iWh,
                      int ibf, int ibg, int ibh) {
    cvt(h, Xb, NEDGE, DIMC, DIMC);
    cvt(F(iWf), Wb, DIMC, DIMC, DIMC);
    gemm(Xb, DIMC, Wb, DIMC, F(ibf), nullptr, t0, NEDGE, DIMC, 0);   // fx
    cvt(F(iWg), Wb, DIMC, DIMC, DIMC);
    gemm(Xb, DIMC, Wb, DIMC, F(ibg), nullptr, t1, NEDGE, DIMC, 3);   // exp
    k_segagg<<<dim3(G, 3), 128, 0, stream>>>(t0, t1, key, gagg, NEDGE, DIMC);
    cvt(gagg, Gb, G, DIMC, DIMC);
    cvt(F(iWh), Wb, DIMC, DIMC, DIMC);
    gemm(Gb, DIMC, Wb, DIMC, F(ibh), nullptr, t2, G, DIMC, 0);       // Wh
    k_gather_add<<<EL, 256, 0, stream>>>(h, t2, key, NEL, DIMC);
  };
  auto gated = [&](const float* x, int iWg, int ibg, int iW1, int iW2,
                   int ib1, int ib2, float* outp) {
    cvt(x, Xb, NEDGE, DIMC, DIMC);
    cvt(F(iWg), Wb, DIMC, DIMC, DIMC);
    gemm(Xb, DIMC, Wb, DIMC, F(ibg), nullptr, t1, NEDGE, DIMC, 2);   // gate
    cvt(F(iW1), Wb, DIMC, DIMC, DIMC);
    gemm(Xb, DIMC, Wb, DIMC, F(ib1), nullptr, t2, NEDGE, DIMC, 1);   // r1
    cvt(t2, Xb, NEDGE, DIMC, DIMC);
    cvt(F(iW2), Wb, DIMC, DIMC, DIMC);
    gemm(Xb, DIMC, Wb, DIMC, F(ib2), nullptr, t2, NEDGE, DIMC, 0);   // r2
    k_grcomb<<<EL, 256, 0, stream>>>(x, t1, t2, outp, NEL);
  };

  // scalar ii bias (sum(ii)*1e-10)
  k_sum_ii<<<1, 256, 0, stream>>>(iip, NEDGE, s_bias);

  // corr MLP: relu(L1) -> L2 -> relu(LN) -> L3
  cvt(corr, Xb, NEDGE, KCORR, KCORR_PAD);
  cvt(F(P_co_W1), Wb, DIMC, KCORR, KCORR_PAD);
  gemm(Xb, KCORR_PAD, Wb, KCORR_PAD, F(P_co_b1), nullptr, t0, NEDGE,
       KCORR_PAD, 1);
  cvt(t0, Xb, NEDGE, DIMC, DIMC);
  cvt(F(P_co_W2), Wb, DIMC, DIMC, DIMC);
  gemm(Xb, DIMC, Wb, DIMC, F(P_co_b2), nullptr, t1, NEDGE, DIMC, 0);
  lnorm(t1, F(P_co_lng), F(P_co_lnb), t0, 1);
  cvt(t0, Xb, NEDGE, DIMC, DIMC);
  cvt(F(P_co_W3), Wb, DIMC, DIMC, DIMC);
  gemm(Xb, DIMC, Wb, DIMC, F(P_co_b3), nullptr, t1, NEDGE, DIMC, 0);

  // h = LN(net + inp + corr_out + bias)
  k_add4<<<EL, 256, 0, stream>>>(net, inp, t1, s_bias, t0, NEL);
  lnorm(t0, F(P_ng), F(P_nb), h, 0);

  // neighbor indices
  k_neighbors<<<16, 256, 0, stream>>>(kkp, jjp, ixb, jxb, NEDGE);

  // h += mlp2(h[ix], c1)
  k_gather_rows<<<EL, 256, 0, stream>>>(h, ixb, t0, NEL, DIMC);
  cvt(t0, Xb, NEDGE, DIMC, DIMC);
  cvt(F(P_c1_W1), Wb, DIMC, DIMC, DIMC);
  gemm(Xb, DIMC, Wb, DIMC, F(P_c1_b1), nullptr, t1, NEDGE, DIMC, 1);
  cvt(t1, Xb, NEDGE, DIMC, DIMC);
  cvt(F(P_c1_W2), Wb, DIMC, DIMC, DIMC);
  gemm(Xb, DIMC, Wb, DIMC, F(P_c1_b2), h, h, NEDGE, DIMC, 0);

  // h += mlp2(h[jx], c2)  (uses updated h)
  k_gather_rows<<<EL, 256, 0, stream>>>(h, jxb, t0, NEL, DIMC);
  cvt(t0, Xb, NEDGE, DIMC, DIMC);
  cvt(F(P_c2_W1), Wb, DIMC, DIMC, DIMC);
  gemm(Xb, DIMC, Wb, DIMC, F(P_c2_b1), nullptr, t1, NEDGE, DIMC, 1);
  cvt(t1, Xb, NEDGE, DIMC, DIMC);
  cvt(F(P_c2_W2), Wb, DIMC, DIMC, DIMC);
  gemm(Xb, DIMC, Wb, DIMC, F(P_c2_b2), h, h, NEDGE, DIMC, 0);

  // scatter-softmax aggregations
  soft_agg(kkp, 768, P_akk_Wf, P_akk_Wg, P_akk_Wh, P_akk_bf, P_akk_bg,
           P_akk_bh);
  k_make_key<<<16, 256, 0, stream>>>(iip, jjp, keyb, NEDGE);
  soft_agg(keyb, 256, P_aij_Wf, P_aij_Wg, P_aij_Wh, P_aij_bf, P_aij_bg,
           P_aij_bh);

  // GRU-style blocks
  lnorm(h, F(P_ln1g), F(P_ln1b), t0, 0);
  gated(t0, P_g1_Wg, P_g1_bg, P_g1_W1, P_g1_W2, P_g1_b1, P_g1_b2, h);
  lnorm(h, F(P_ln2g), F(P_ln2b), t0, 0);
  float* out_h = (float*)d_out;
  gated(t0, P_g2_Wg, P_g2_bg, P_g2_W1, P_g2_W2, P_g2_b1, P_g2_b2, out_h);

  // heads (h -> d, w) written after h region in d_out
  float* outd = out_h + (size_t)NEDGE * DIMC;
  float* outw = outd + (size_t)NEDGE * 2;
  k_heads<<<NEDGE, 128, 0, stream>>>(out_h, F(P_Wd), F(P_bd), F(P_Ww),
                                     F(P_bw), iip, outd, outw);
}
